// DeformConv_experimental_1881195676322
// MI455X (gfx1250) — matrix operand order repaired and checked by reading, not compiled, not run on hardware
//
#include <hip/hip_runtime.h>

typedef __attribute__((ext_vector_type(16))) _Float16 v16h;
typedef __attribute__((ext_vector_type(8)))  float    v8f;

// Problem constants (from setup_inputs)
#define NB   8      // batch
#define CIN  256
#define HH   64
#define WW   64
#define HW   4096   // 64*64
#define OFFC 18
#define COUT 256
#define KTAP 9      // 3x3
// GEMM: M = NB*HW = 32768, Ncols = COUT = 256, K = CIN*KTAP = 2304
#define NCHUNK 72   // 2304 / 32
#define NTILES 16   // 256 / 16

// ---------------------------------------------------------------------------
// Kernel 0: transpose x (N,C,H,W) -> xt (N,H,W,C) so bilinear gathers read
// contiguous channel runs (64B per lane per corner instead of 16KB strides).
// ---------------------------------------------------------------------------
__global__ void k_transpose(const float* __restrict__ x, float* __restrict__ xt) {
    __shared__ float tile[32][33];
    const int b  = blockIdx.x;            // (n, h, cb, wb): 8*64*8*2 = 8192
    const int wb = (b & 1) * 32;          // W/32 = 2
    const int cb = ((b >> 1) & 7) * 32;   // C/32 = 8
    const int h  = (b >> 4) & 63;
    const int n  = b >> 10;
    const int tx = threadIdx.x & 31;
    const int ty = threadIdx.x >> 5;      // 0..7

    const float* src = x + (size_t)n * CIN * HW + h * WW;
    #pragma unroll
    for (int r = 0; r < 32; r += 8)
        tile[ty + r][tx] = src[(size_t)(cb + ty + r) * HW + (wb + tx)];
    __syncthreads();

    float* dst = xt + ((size_t)n * HW + (size_t)h * WW) * CIN;
    #pragma unroll
    for (int r = 0; r < 32; r += 8)
        dst[(size_t)(wb + ty + r) * CIN + (cb + tx)] = tile[tx][ty + r];
}

// ---------------------------------------------------------------------------
// Kernel 1: 1x1 conv  x_chan[n,oc,h,w] = b_adj[oc] + sum_c x[n,c,h,w]*w_adj[oc,c]
// ---------------------------------------------------------------------------
__global__ void k_conv1x1(const float* __restrict__ x,
                          const float* __restrict__ wadj,
                          const float* __restrict__ badj,
                          float* __restrict__ xchan) {
    __shared__ float wsh[OFFC * CIN];
    for (int i = threadIdx.x; i < OFFC * CIN; i += blockDim.x) wsh[i] = wadj[i];
    __syncthreads();

    int t = blockIdx.x * blockDim.x + threadIdx.x;   // 0 .. NB*HW-1
    int n = t >> 12;
    int hw = t & (HW - 1);
    const float* xp = x + (size_t)n * CIN * HW + hw;

    float acc[OFFC];
    #pragma unroll
    for (int oc = 0; oc < OFFC; ++oc) acc[oc] = badj[oc];

    for (int c = 0; c < CIN; ++c) {
        float xv = xp[(size_t)c * HW];
        #pragma unroll
        for (int oc = 0; oc < OFFC; ++oc) acc[oc] += xv * wsh[oc * CIN + c];
    }
    #pragma unroll
    for (int oc = 0; oc < OFFC; ++oc)
        xchan[((size_t)n * OFFC + oc) * HW + hw] = acc[oc];
}

// ---------------------------------------------------------------------------
// Kernel 2: depthwise 3x3 (groups=18, pad=1)
// ---------------------------------------------------------------------------
__global__ void k_dw3x3(const float* __restrict__ xchan,
                        const float* __restrict__ woff,
                        const float* __restrict__ boff,
                        float* __restrict__ offs) {
    int t = blockIdx.x * blockDim.x + threadIdx.x;   // 0 .. NB*OFFC*HW-1
    int hw = t & (HW - 1);
    int w  = hw & (WW - 1);
    int h  = hw >> 6;
    int rest = t >> 12;
    int oc = rest % OFFC;
    int n  = rest / OFFC;

    const float* src = xchan + ((size_t)n * OFFC + oc) * HW;
    float a = boff[oc];
    #pragma unroll
    for (int i = 0; i < 3; ++i) {
        #pragma unroll
        for (int j = 0; j < 3; ++j) {
            int hy = h - 1 + i;
            int wx = w - 1 + j;
            bool ok = (hy >= 0) & (hy < HH) & (wx >= 0) & (wx < WW);
            float v = ok ? src[hy * WW + wx] : 0.0f;
            a += v * woff[oc * 9 + i * 3 + j];
        }
    }
    offs[t] = a;
}

// ---------------------------------------------------------------------------
// Kernel 3: pack w_def (Cout,C,3,3) f32 -> f16 WMMA B fragments.
// K flattening kk = kpos*256 + c ; chunk = kpos*8 + cblk holds c = cblk*32+kidx.
// B fragment (16x16x32 f16, wave32): element j of lane l holds B[k][l%16], h = l/16,
// k = 8h + j (j<8), 16 + 8h + (j-8) (j>=8); same k order as the A gather.
// ---------------------------------------------------------------------------
__global__ void k_pack_b(const float* __restrict__ wdef, v16h* __restrict__ pb) {
    int t = blockIdx.x * blockDim.x + threadIdx.x;   // 72*16*32 = 36864
    if (t >= NCHUNK * NTILES * 32) return;
    int chunk = t >> 9;
    int r     = t & 511;
    int lane  = r & 31;
    int kpos  = chunk >> 3;
    int cblk  = chunk & 7;
    int o     = ((r >> 5) << 4) + (lane & 15);
    int khalf = (lane & 16) ? 16 : 0;

    v16h frag;
    #pragma unroll
    for (int j = 0; j < 16; ++j) {
        int c = cblk * 32 + (j & 7) + (khalf >> 1) + ((j & 8) << 1);
        frag[j] = (_Float16)wdef[((size_t)o * CIN + c) * KTAP + kpos];
    }
    pb[t] = frag;
}

// ---------------------------------------------------------------------------
// Bilinear blend of four float4 corner vectors
// ---------------------------------------------------------------------------
__device__ __forceinline__ float4 bilerp4(float4 a, float4 b, float4 c, float4 d,
                                          float w00, float w01, float w10, float w11) {
    float4 r;
    r.x = w00 * a.x + w01 * b.x + w10 * c.x + w11 * d.x;
    r.y = w00 * a.y + w01 * b.y + w10 * c.y + w11 * d.y;
    r.z = w00 * a.z + w01 * b.z + w10 * c.z + w11 * d.z;
    r.w = w00 * a.w + w01 * b.w + w10 * c.w + w11 * d.w;
    return r;
}

// ---------------------------------------------------------------------------
// Kernel 4: deformable conv as WMMA GEMM, NHWC gathers + double-buffered B
// with software-pipelined B-fragment reads.
// Wave -> one M-tile of 16 pixels, all 16 Cout tiles (128 accum VGPRs).
// ---------------------------------------------------------------------------
__global__ void __launch_bounds__(256, 1)
k_deform_wmma(const float* __restrict__ xt,
              const float* __restrict__ offs,
              const v16h* __restrict__ pb,
              float* __restrict__ out) {
    __shared__ v16h bsh[2][NTILES * 32];   // 2 x 16 KB double buffer

    const int lane  = threadIdx.x & 31;
    const int wv    = threadIdx.x >> 5;
    const int mtile = blockIdx.x * 8 + wv;          // 0..2047
    const int pbase = mtile << 4;
    const int n     = pbase >> 12;
    const int rem   = pbase & (HW - 1);
    const int ho    = rem >> 6;
    const int wob   = rem & (WW - 1);               // multiple of 16
    const int mrow  = lane & 15;
    const int wo    = wob + mrow;
    const int hihalf = (lane >> 4) & 1;
    const int cshift = hihalf ? 8 : 0;

    const float* xtn = xt + (size_t)n * HW * CIN;   // [h][w][c]

    v8f acc[NTILES];
    #pragma unroll
    for (int nt = 0; nt < NTILES; ++nt) acc[nt] = (v8f)0.0f;

    // prologue: stage chunk 0's B fragments into buffer 0
    bsh[0][threadIdx.x]       = pb[threadIdx.x];
    bsh[0][threadIdx.x + 256] = pb[threadIdx.x + 256];

    // bilinear corner state (refreshed every 8 chunks)
    float w00 = 0.f, w01 = 0.f, w10 = 0.f, w11 = 0.f;
    const float *q00 = xtn, *q01 = xtn, *q10 = xtn, *q11 = xtn;

    for (int chunk = 0; chunk < NCHUNK; ++chunk) {
        const int kpos = chunk >> 3;
        const int cblk = chunk & 7;

        if (cblk == 0) {   // uniform branch: per-tap bilinear setup
            const float offy = offs[(((size_t)n * OFFC + 2 * kpos)     << 12) + (ho << 6) + wo];
            const float offx = offs[(((size_t)n * OFFC + 2 * kpos + 1) << 12) + (ho << 6) + wo];
            const float py = (float)(ho - 1 + kpos / 3) + offy;
            const float px = (float)(wo - 1 + kpos % 3) + offx;
            const float fy = floorf(py), fx = floorf(px);
            const float wy = py - fy,    wx = px - fx;
            const int y0 = (int)fy, x0 = (int)fx;
            const int y1 = y0 + 1,  x1 = x0 + 1;
            const bool vy0 = (y0 >= 0) & (y0 < HH);
            const bool vy1 = (y1 >= 0) & (y1 < HH);
            const bool vx0 = (x0 >= 0) & (x0 < WW);
            const bool vx1 = (x1 >= 0) & (x1 < WW);
            const int cy0 = min(max(y0, 0), HH - 1) * WW;
            const int cy1 = min(max(y1, 0), HH - 1) * WW;
            const int cx0 = min(max(x0, 0), WW - 1);
            const int cx1 = min(max(x1, 0), WW - 1);
            q00 = xtn + (size_t)(cy0 + cx0) * CIN;
            q01 = xtn + (size_t)(cy0 + cx1) * CIN;
            q10 = xtn + (size_t)(cy1 + cx0) * CIN;
            q11 = xtn + (size_t)(cy1 + cx1) * CIN;
            w00 = (vy0 & vx0) ? (1.0f - wy) * (1.0f - wx) : 0.0f;
            w01 = (vy0 & vx1) ? (1.0f - wy) * wx          : 0.0f;
            w10 = (vy1 & vx0) ? wy * (1.0f - wx)          : 0.0f;
            w11 = (vy1 & vx1) ? wy * wx                   : 0.0f;
        }

        __syncthreads();   // chunk's B staged & previous reads of both bufs done
        const int buf = chunk & 1;
        const v16h* bbase = bsh[buf];

        // 1) start the B-fragment read pipeline (these DS loads precede the
        //    staging stores in issue order, so WMMA waits don't drain staging)
        v16h bf0 = bbase[lane];
        v16h bf1 = bbase[32 + lane];

        // 2) issue all 16 A-gather b128 loads as early as possible
        const int cbase = cblk * 32 + cshift;
        const float4* p00 = (const float4*)(q00 + cbase);
        const float4* p01 = (const float4*)(q01 + cbase);
        const float4* p10 = (const float4*)(q10 + cbase);
        const float4* p11 = (const float4*)(q11 + cbase);
        float4 a0 = p00[0], a1 = p00[1], a2 = p00[4], a3 = p00[5];
        float4 b0 = p01[0], b1 = p01[1], b2 = p01[4], b3 = p01[5];
        float4 c0 = p10[0], c1 = p10[1], c2 = p10[4], c3 = p10[5];
        float4 d0 = p11[0], d1 = p11[1], d2 = p11[4], d3 = p11[5];

        // 3) stage NEXT chunk's B into the other buffer (overlaps compute)
        if (chunk + 1 < NCHUNK) {
            const v16h* src = pb + (size_t)(chunk + 1) * (NTILES * 32);
            bsh[buf ^ 1][threadIdx.x]       = src[threadIdx.x];
            bsh[buf ^ 1][threadIdx.x + 256] = src[threadIdx.x + 256];
        }

        // 4) bilinear blend + f16 pack (channels cbase+{0..7}, cbase+{16..23})
        float4 g0 = bilerp4(a0, b0, c0, d0, w00, w01, w10, w11);
        float4 g1 = bilerp4(a1, b1, c1, d1, w00, w01, w10, w11);
        float4 g2 = bilerp4(a2, b2, c2, d2, w00, w01, w10, w11);
        float4 g3 = bilerp4(a3, b3, c3, d3, w00, w01, w10, w11);

        v16h afrag;
        afrag[0]  = (_Float16)g0.x; afrag[1]  = (_Float16)g0.y;
        afrag[2]  = (_Float16)g0.z; afrag[3]  = (_Float16)g0.w;
        afrag[4]  = (_Float16)g1.x; afrag[5]  = (_Float16)g1.y;
        afrag[6]  = (_Float16)g1.z; afrag[7]  = (_Float16)g1.w;
        afrag[8]  = (_Float16)g2.x; afrag[9]  = (_Float16)g2.y;
        afrag[10] = (_Float16)g2.z; afrag[11] = (_Float16)g2.w;
        afrag[12] = (_Float16)g3.x; afrag[13] = (_Float16)g3.y;
        afrag[14] = (_Float16)g3.z; afrag[15] = (_Float16)g3.w;

        // 5) 16 WMMAs with B reads pipelined 2 tiles ahead
        #pragma unroll
        for (int nt = 0; nt < NTILES; ++nt) {
            v16h bnext = bf1;
            if (nt + 2 < NTILES) bnext = bbase[(nt + 2) * 32 + lane];
            acc[nt] = __builtin_amdgcn_wmma_f32_16x16x32_f16(
                false, afrag, false, bf0, (short)0, acc[nt], false, false);
            bf0 = bf1;
            bf1 = bnext;
        }
    }

    // ---- epilogue: D VGPR r holds M = r + 8*hihalf, N = lane&15.
    float* orow = out + (size_t)n * COUT * HW + (ho << 6) + wob + hihalf * 8;
    #pragma unroll
    for (int nt = 0; nt < NTILES; ++nt) {
        const int o = nt * 16 + mrow;
        float* dst = orow + (size_t)o * HW;
        float4 lo = make_float4(acc[nt][0], acc[nt][1], acc[nt][2], acc[nt][3]);
        float4 hi = make_float4(acc[nt][4], acc[nt][5], acc[nt][6], acc[nt][7]);
        *(float4*)(dst)     = lo;
        *(float4*)(dst + 4) = hi;
    }
}

// ---------------------------------------------------------------------------
// Host-side launcher (graph-capture safe: only kernel launches on stream)
// ---------------------------------------------------------------------------
extern "C" void kernel_launch(void* const* d_in, const int* in_sizes, int n_in,
                              void* d_out, int out_size, void* d_ws, size_t ws_size,
                              hipStream_t stream) {
    const float* x     = (const float*)d_in[0];
    const float* w_adj = (const float*)d_in[1];
    const float* b_adj = (const float*)d_in[2];
    const float* w_off = (const float*)d_in[3];
    const float* b_off = (const float*)d_in[4];
    const float* w_def = (const float*)d_in[5];
    float* out = (float*)d_out;

    // Workspace layout (floats): xt (NHWC) | xchan | offs | packed B
    float* ws    = (float*)d_ws;
    float* xt    = ws;                                         // 8*4096*256
    float* xchan = xt + (size_t)NB * HW * CIN;                 // 8*18*4096
    float* offs  = xchan + (size_t)NB * OFFC * HW;             // 8*18*4096
    v16h*  pb    = (v16h*)(offs + (size_t)NB * OFFC * HW);     // 72*16*32 v16h

    k_transpose<<<NB * HH * (CIN / 32) * (WW / 32), 256, 0, stream>>>(x, xt);
    k_conv1x1<<<(NB * HW) / 256, 256, 0, stream>>>(x, w_adj, b_adj, xchan);
    k_dw3x3  <<<(NB * OFFC * HW) / 256, 256, 0, stream>>>(xchan, w_off, b_off, offs);
    k_pack_b <<<(NCHUNK * NTILES * 32 + 255) / 256, 256, 0, stream>>>(w_def, pb);
    k_deform_wmma<<<256, 256, 0, stream>>>(xt, offs, pb, out);
}
